// ContinuousFilterConvolution_58325655879889
// MI455X (gfx1250) — matrix operand order repaired and checked by reading, not compiled, not run on hardware
//
#include <hip/hip_runtime.h>
#include <hip/hip_bf16.h>
#include <math.h>

// ---------------------------------------------------------------------------
// Types for CDNA5 WMMA (wave32)
// ---------------------------------------------------------------------------
typedef __bf16 bf16t;
typedef __attribute__((ext_vector_type(16))) __bf16 v16bf;
typedef __attribute__((ext_vector_type(8)))  __bf16 v8bf;
typedef __attribute__((ext_vector_type(8)))  float  v8f;

// ---------------------------------------------------------------------------
// Problem constants (match reference)
// ---------------------------------------------------------------------------
constexpr int CH      = 256;      // channels
constexpr int NCEN    = 200;      // rbf centers
constexpr int KP      = 224;      // rbf K padded to multiple of 32
constexpr int NNODES  = 20000;
constexpr int NEDGES  = 320000;
constexpr int TE      = 64;       // edges per block
constexpr int TN      = 32;       // nodes per block
constexpr int RBSTR   = KP + 8;   // LDS row stride (bf16 elems), 16B-aligned rows
constexpr int HS      = CH + 8;   // LDS row stride for 256-wide activations
constexpr float CSTEP = 20.0f / 199.0f;  // linspace(0,20,200) spacing
constexpr float LOG2F_ = 0.69314718055994531f;

// Workspace layout (bytes)
constexpr size_t O_W1T  = 0;                         // [256][224] bf16 (transposed)
constexpr size_t O_W2T  = O_W1T + 256ull*KP*2;       // [256][256] bf16 (transposed)
constexpr size_t O_W3T  = O_W2T + 256ull*256*2;
constexpr size_t O_W4T  = O_W3T + 256ull*256*2;
constexpr size_t O_W5T  = O_W4T + 256ull*256*2;
constexpr size_t O_XW3  = O_W5T + 256ull*256*2;      // [N][256] f32
constexpr size_t O_CONV = O_XW3 + (size_t)NNODES*CH*4;

// ---------------------------------------------------------------------------
// Helpers
// ---------------------------------------------------------------------------
__device__ __forceinline__ float sspf(float x) {
    float sp = (x < 14.f) ? __logf(1.f + __expf(x)) : x;
    return sp - LOG2F_;
}

__device__ __forceinline__ v8f wmma_bf16(v16bf a, v16bf b, v8f c) {
    // v_wmma_f32_16x16x32_bf16: (neg_a, A, neg_b, B, c_mod, C, reuse_a, reuse_b)
    return __builtin_amdgcn_wmma_f32_16x16x32_bf16(false, a, false, b,
                                                   (short)0, c, false, false);
}

// A operand (16x32 bf16): lane L -> row M=L%16; element j -> K = half*8+j (j<8),
// 16+half*8+(j-8) (j>=8).  Two contiguous 16B chunks from an LDS row.
__device__ __forceinline__ v16bf load_a(const bf16t* base, int stride,
                                        int m, int kb, int half) {
    const v8bf* p0 = (const v8bf*)(base + m * stride + kb + half * 8);
    const v8bf* p1 = (const v8bf*)(base + m * stride + kb + 16 + half * 8);
    v8bf lo = *p0, hi = *p1;
    return __builtin_shufflevector(lo, hi, 0,1,2,3,4,5,6,7, 8,9,10,11,12,13,14,15);
}

// B operand (32x16 bf16): lane L -> col N=L%16; element j -> K = half*8+j (j<8),
// 16+half*8+(j-8) (j>=8).  Weight stored transposed Wt[N][K]: two 16B chunks.
__device__ __forceinline__ v16bf load_b(const bf16t* wt, int wstride,
                                        int n, int kb, int half) {
    const v8bf* p0 = (const v8bf*)(wt + (size_t)n * wstride + kb + half * 8);
    const v8bf* p1 = (const v8bf*)(wt + (size_t)n * wstride + kb + 16 + half * 8);
    v8bf lo = *p0, hi = *p1;
    return __builtin_shufflevector(lo, hi, 0,1,2,3,4,5,6,7, 8,9,10,11,12,13,14,15);
}

// Joint register block: 2 column tiles (n0, n0+16) x R row tiles.
// Per K-step: 2 B loads + R A loads -> 2R WMMAs (A and B each loaded once).
template<int R>
__device__ __forceinline__ void wave_gemm_joint(const bf16t* __restrict__ A, int astr,
                                                const bf16t* __restrict__ Wt, int wstr,
                                                int kTot, int n0, int ln, int half,
                                                v8f acc[2][R]) {
    for (int kb = 0; kb < kTot; kb += 32) {
        v16bf b0 = load_b(Wt, wstr, n0,      kb, half);
        v16bf b1 = load_b(Wt, wstr, n0 + 16, kb, half);
        #pragma unroll
        for (int r = 0; r < R; ++r) {
            v16bf a = load_a(A, astr, r * 16 + ln, kb, half);
            acc[0][r] = wmma_bf16(a, b0, acc[0][r]);
            acc[1][r] = wmma_bf16(a, b1, acc[1][r]);
        }
    }
}

template<int R>
__device__ __forceinline__ void zero_acc(v8f acc[2][R]) {
    v8f z = {};
    #pragma unroll
    for (int c = 0; c < 2; ++c)
        #pragma unroll
        for (int r = 0; r < R; ++r) acc[c][r] = z;
}

// ---------------------------------------------------------------------------
// Kernel: convert + transpose weights to bf16 (w1 zero-padded K 200 -> 224)
// ---------------------------------------------------------------------------
__global__ void k_prep(const float* __restrict__ w1, const float* __restrict__ w2,
                       const float* __restrict__ w3, const float* __restrict__ w4,
                       const float* __restrict__ w5,
                       bf16t* __restrict__ w1t, bf16t* __restrict__ w2t,
                       bf16t* __restrict__ w3t, bf16t* __restrict__ w4t,
                       bf16t* __restrict__ w5t) {
    int tid = blockIdx.x * blockDim.x + threadIdx.x;
    int stride = gridDim.x * blockDim.x;
    for (int i = tid; i < 256 * KP; i += stride) {
        int n = i / KP, k = i - n * KP;
        float v = (k < NCEN) ? w1[k * CH + n] : 0.f;
        w1t[i] = (bf16t)v;
    }
    for (int i = tid; i < CH * CH; i += stride) {
        int n = i >> 8, k = i & 255;
        w2t[i] = (bf16t)w2[k * CH + n];
        w3t[i] = (bf16t)w3[k * CH + n];
        w4t[i] = (bf16t)w4[k * CH + n];
        w5t[i] = (bf16t)w5[k * CH + n];
    }
}

__global__ void k_zero(float* __restrict__ p, int n) {
    int tid = blockIdx.x * blockDim.x + threadIdx.x;
    for (int i = tid; i < n; i += gridDim.x * blockDim.x) p[i] = 0.f;
}

// ---------------------------------------------------------------------------
// Kernel: xw3 = x @ w3   (625 blocks x 32 rows, 8 waves: 2 col-tiles each)
// ---------------------------------------------------------------------------
__global__ void __launch_bounds__(256)
k_xw3(const float* __restrict__ x, const bf16t* __restrict__ w3t,
      float* __restrict__ xw3) {
    __shared__ bf16t xa[TN * HS];
    const int tid = threadIdx.x;
    const int base = blockIdx.x * TN;
    for (int i = tid; i < TN * CH; i += 256) {
        int m = i >> 8, k = i & 255;
        xa[m * HS + k] = (bf16t)x[(size_t)(base + m) * CH + k];
    }
    __syncthreads();
    const int wid = tid >> 5, lane = tid & 31, half = lane >> 4, ln = lane & 15;
    const int n0 = wid * 32 + ln;
    v8f acc[2][2];
    zero_acc<2>(acc);
    wave_gemm_joint<2>(xa, HS, w3t, CH, CH, n0, ln, half, acc);
    #pragma unroll
    for (int c = 0; c < 2; ++c) {
        int n = n0 + c * 16;
        #pragma unroll
        for (int r = 0; r < 2; ++r)
            #pragma unroll
            for (int q = 0; q < 8; ++q)
                xw3[(size_t)(base + r * 16 + half * 8 + q) * CH + n] = acc[c][r][q];
    }
}

// ---------------------------------------------------------------------------
// Kernel: fused edge pipeline: dist -> rbf -> ssp(rbf@w1+b1) -> ssp(h@w2+b2)
//         -> w * xw3[sender] -> atomic scatter-add into conv[receiver]
// 5000 blocks x 64 edges, 256 threads (8 waves, 2 col-tiles x 4 row-tiles each)
// ---------------------------------------------------------------------------
__global__ void __launch_bounds__(256)
k_edge(const float* __restrict__ pos, const int* __restrict__ senders,
       const int* __restrict__ receivers,
       const bf16t* __restrict__ w1t, const float* __restrict__ b1,
       const bf16t* __restrict__ w2t, const float* __restrict__ b2,
       const float* __restrict__ xw3, float* __restrict__ conv) {
    __shared__ int   sS[TE];
    __shared__ int   sR[TE];
    __shared__ float sD[TE];
    __shared__ bf16t rbfa[TE * RBSTR];
    __shared__ bf16t ha[TE * HS];
    __shared__ bf16t wa[TE * HS];

    const int tid = threadIdx.x;
    const int e0 = blockIdx.x * TE;
    if (tid < TE) {
        int e = e0 + tid;
        int s = senders[e], r = receivers[e];
        sS[tid] = s; sR[tid] = r;
        float dx = pos[3 * r + 0] - pos[3 * s + 0];
        float dy = pos[3 * r + 1] - pos[3 * s + 1];
        float dz = pos[3 * r + 2] - pos[3 * s + 2];
        sD[tid] = sqrtf(dx * dx + dy * dy + dz * dz);
    }
    __syncthreads();
    // RBF expansion (padded cols 200..223 = 0)
    for (int i = tid; i < TE * KP; i += 256) {
        int e = i / KP, k = i - e * KP;
        float v = 0.f;
        if (k < NCEN) {
            float d = sD[e] - CSTEP * (float)k;
            v = __expf(-d * d * 10.0f);   // exp(-(d-c)^2 / 0.1)
        }
        rbfa[e * RBSTR + k] = (bf16t)v;
    }
    __syncthreads();

    const int wid = tid >> 5, lane = tid & 31, half = lane >> 4, ln = lane & 15;
    const int n0 = wid * 32 + ln;
    // GEMM1: h = ssp(rbf @ w1 + b1)
    {
        v8f acc[2][4];
        zero_acc<4>(acc);
        wave_gemm_joint<4>(rbfa, RBSTR, w1t, KP, KP, n0, ln, half, acc);
        #pragma unroll
        for (int c = 0; c < 2; ++c) {
            int n = n0 + c * 16;
            float bias = b1[n];
            #pragma unroll
            for (int r = 0; r < 4; ++r)
                #pragma unroll
                for (int q = 0; q < 8; ++q)
                    ha[(r * 16 + half * 8 + q) * HS + n] = (bf16t)sspf(acc[c][r][q] + bias);
        }
    }
    __syncthreads();
    // GEMM2: w = ssp(h @ w2 + b2)
    {
        v8f acc[2][4];
        zero_acc<4>(acc);
        wave_gemm_joint<4>(ha, HS, w2t, CH, CH, n0, ln, half, acc);
        #pragma unroll
        for (int c = 0; c < 2; ++c) {
            int n = n0 + c * 16;
            float bias = b2[n];
            #pragma unroll
            for (int r = 0; r < 4; ++r)
                #pragma unroll
                for (int q = 0; q < 8; ++q)
                    wa[(r * 16 + half * 8 + q) * HS + n] = (bf16t)sspf(acc[c][r][q] + bias);
        }
    }
    __syncthreads();
    // wf = w * xw3[sender], scatter-add to conv[receiver] (f32 HW atomics, L2)
    for (int e = 0; e < TE; ++e) {
        float f  = xw3[(size_t)sS[e] * CH + tid];
        float wv = (float)wa[e * HS + tid];
        __hip_atomic_fetch_add(&conv[(size_t)sR[e] * CH + tid], wv * f,
                               __ATOMIC_RELAXED, __HIP_MEMORY_SCOPE_AGENT);
    }
}

// ---------------------------------------------------------------------------
// Kernel: out = x + (ssp(conv @ w4 + b4) @ w5 + b5)    (625 blocks x 32 rows)
// ---------------------------------------------------------------------------
__global__ void __launch_bounds__(256)
k_node(const float* __restrict__ conv, const float* __restrict__ x,
       const bf16t* __restrict__ w4t, const float* __restrict__ b4,
       const bf16t* __restrict__ w5t, const float* __restrict__ b5,
       float* __restrict__ out) {
    __shared__ bf16t ca[TN * HS];
    __shared__ bf16t ya[TN * HS];
    const int tid = threadIdx.x;
    const int base = blockIdx.x * TN;
    for (int i = tid; i < TN * CH; i += 256) {
        int m = i >> 8, k = i & 255;
        ca[m * HS + k] = (bf16t)conv[(size_t)(base + m) * CH + k];
    }
    __syncthreads();
    const int wid = tid >> 5, lane = tid & 31, half = lane >> 4, ln = lane & 15;
    const int n0 = wid * 32 + ln;
    // y = ssp(conv @ w4 + b4)
    {
        v8f acc[2][2];
        zero_acc<2>(acc);
        wave_gemm_joint<2>(ca, HS, w4t, CH, CH, n0, ln, half, acc);
        #pragma unroll
        for (int c = 0; c < 2; ++c) {
            int n = n0 + c * 16;
            float bias = b4[n];
            #pragma unroll
            for (int r = 0; r < 2; ++r)
                #pragma unroll
                for (int q = 0; q < 8; ++q)
                    ya[(r * 16 + half * 8 + q) * HS + n] = (bf16t)sspf(acc[c][r][q] + bias);
        }
    }
    __syncthreads();
    // out = x + y @ w5 + b5
    {
        v8f acc[2][2];
        zero_acc<2>(acc);
        wave_gemm_joint<2>(ya, HS, w5t, CH, CH, n0, ln, half, acc);
        #pragma unroll
        for (int c = 0; c < 2; ++c) {
            int n = n0 + c * 16;
            float bias = b5[n];
            #pragma unroll
            for (int r = 0; r < 2; ++r)
                #pragma unroll
                for (int q = 0; q < 8; ++q) {
                    int m = base + r * 16 + half * 8 + q;
                    out[(size_t)m * CH + n] = x[(size_t)m * CH + n] + acc[c][r][q] + bias;
                }
        }
    }
}

// ---------------------------------------------------------------------------
// Launch
// ---------------------------------------------------------------------------
extern "C" void kernel_launch(void* const* d_in, const int* in_sizes, int n_in,
                              void* d_out, int out_size, void* d_ws, size_t ws_size,
                              hipStream_t stream) {
    const float* pos = (const float*)d_in[0];
    const float* x   = (const float*)d_in[1];
    const int*   snd = (const int*)d_in[2];
    const int*   rcv = (const int*)d_in[3];
    const float* w1  = (const float*)d_in[4];
    const float* b1  = (const float*)d_in[5];
    const float* w2  = (const float*)d_in[6];
    const float* b2  = (const float*)d_in[7];
    const float* w3  = (const float*)d_in[8];
    const float* w4  = (const float*)d_in[9];
    const float* b4  = (const float*)d_in[10];
    const float* w5  = (const float*)d_in[11];
    const float* b5  = (const float*)d_in[12];

    char* ws = (char*)d_ws;
    bf16t* w1t = (bf16t*)(ws + O_W1T);
    bf16t* w2t = (bf16t*)(ws + O_W2T);
    bf16t* w3t = (bf16t*)(ws + O_W3T);
    bf16t* w4t = (bf16t*)(ws + O_W4T);
    bf16t* w5t = (bf16t*)(ws + O_W5T);
    float* xw3  = (float*)(ws + O_XW3);
    float* conv = (float*)(ws + O_CONV);

    k_prep<<<256, 256, 0, stream>>>(w1, w2, w3, w4, w5, w1t, w2t, w3t, w4t, w5t);
    k_zero<<<1024, 256, 0, stream>>>(conv, NNODES * CH);
    k_xw3<<<NNODES / TN, 256, 0, stream>>>(x, w3t, xw3);
    k_edge<<<NEDGES / TE, 256, 0, stream>>>(pos, snd, rcv, w1t, b1, w2t, b2, xw3, conv);
    k_node<<<NNODES / TN, 256, 0, stream>>>(conv, x, w4t, b4, w5t, b5, (float*)d_out);
}
